// HawkesHyperGraphLayer_14448269984336
// MI455X (gfx1250) — compile-verified, matrix operand order checked by reading, not run on hardware
//
#include <hip/hip_runtime.h>
#include <hip/hip_bf16.h>
#include <math.h>

// ---------------------------------------------------------------------------
// HawkesHyperGraphLayer on gfx1250 (MI455X).
//
// Algebraic simplification (exact up to fp reassociation):
//   scores[b,l] = leaky_relu( Vt[b,l,:]·u_node + (E_feat[b,:]·u_edge), 0.2 )
// with u_node = W_node^T a_node, u_edge = W_edge^T a_edge.
// This deletes the 215-GFLOP (B·L,H)x(H,H) GEMM; the kernel becomes
// gather-bound (~839MB of embedding rows -> ~36us at 23.3TB/s; less after
// skipping rows past each sequence length).
//
// Per-block (1 block = 1 batch row, 256 thr = 8 waves):
//   A) cumsum(edge_times) -> ts, Hawkes weights w[l] (LDS)
//   B) gather V rows once, scale by w[l], store f16 tile Vt (208x512) in LDS
//      (208KB -- uses CDNA5's 320KB/WGP LDS; rows l>=len are zero, no gather)
//   C) scores via v_wmma_f32_16x16x32_f16: A = 16 Vt rows, B = u_node
//      broadcast into all 16 columns (K-sliced, 16 WMMA per 16-row tile)
//   D) E_feat·u_edge via LDS reduction + ds_add_f32
//   E) masked leaky-relu softmax (wave0, shfl_xor wave32 reductions)
//   F) out[b,h] = sum_l att[l]*Vt[l,h] from LDS
// ---------------------------------------------------------------------------

#define HDIM   512
#define LSEQ   200
#define LPAD   208           // 13 tiles of 16 rows
#define NTILES 13

// LDS layout (bytes), all 16B aligned
#define OFF_VT   0                         // 208*512*2 = 212992
#define OFF_W    212992                    // 208*4
#define OFF_TS   213824                    // 208*4
#define OFF_SC   214656                    // 208*4
#define OFF_ATT  215488                    // 208*4
#define OFF_UN   216320                    // 512*2
#define OFF_MISC 217344                    // 16*4
#define SMEM_BYTES 217408

typedef __attribute__((ext_vector_type(16))) _Float16 v16h;
typedef __attribute__((ext_vector_type(8)))  _Float16 v8h;
typedef __attribute__((ext_vector_type(4)))  _Float16 v4h;
typedef __attribute__((ext_vector_type(8)))  float    v8f;

// --- Kernel 1: u_node = W_node^T a_node ; u_edge = W_edge^T a_edge ----------
__global__ void hawkes_prep_u(const float* __restrict__ Wn,
                              const float* __restrict__ We,
                              const float* __restrict__ av,
                              float* __restrict__ u_edge,
                              _Float16* __restrict__ u_node_h)
{
    int h = blockIdx.x * blockDim.x + threadIdx.x;
    if (h >= HDIM) return;
    float an = 0.f, ae = 0.f;
    for (int o = 0; o < HDIM; ++o) {
        an = fmaf(av[o],        Wn[o * HDIM + h], an);
        ae = fmaf(av[HDIM + o], We[o * HDIM + h], ae);
    }
    u_edge[h]   = ae;
    u_node_h[h] = (_Float16)an;
}

// --- Kernel 2: one block per batch row --------------------------------------
__global__ __launch_bounds__(256, 1)
void hawkes_block_kernel(const float* __restrict__ poi,
                         const int*   __restrict__ seqs,
                         const int*   __restrict__ lengths,
                         const float* __restrict__ edges,
                         const float* __restrict__ u_edge,
                         const _Float16* __restrict__ u_node_h,
                         float* __restrict__ out)
{
    extern __shared__ char smem[];
    _Float16* vt   = (_Float16*)(smem + OFF_VT);
    float*    wv   = (float*)   (smem + OFF_W);
    float*    ts   = (float*)   (smem + OFF_TS);
    float*    sc   = (float*)   (smem + OFF_SC);
    float*    att  = (float*)   (smem + OFF_ATT);
    _Float16* un   = (_Float16*)(smem + OFF_UN);
    float*    misc = (float*)   (smem + OFF_MISC);

    const int tid  = threadIdx.x;
    const int wid  = tid >> 5;
    const int lane = tid & 31;
    const int b    = blockIdx.x;
    const int len  = lengths[b];

    // ---- Phase A: weights/cumsum -------------------------------------------
    for (int i = tid; i < HDIM; i += 256) un[i] = u_node_h[i];
    if (tid == 0) { ts[0] = 0.f; misc[1] = 0.f; }
    if (tid < LSEQ - 1)
        ts[tid + 1] = (tid < len - 1) ? edges[b * (LSEQ - 1) + tid] : 0.f;
    __syncthreads();
    if (tid == 0) {
        float r = 0.f;
        for (int i = 1; i < LSEQ; ++i) { r += ts[i]; ts[i] = r; }
        misc[0] = ts[len - 1];                       // t_last
    }
    __syncthreads();
    const float t_last = misc[0];
    for (int l = tid; l < LPAD; l += 256)
        wv[l] = (l < len) ? __expf(-(t_last - ts[l])) : 0.f;  // GAMMA = 1
    __syncthreads();

    // ---- Phase B: gather rows once, scale, stash f16 tile in LDS -----------
    for (int l = wid; l < LPAD; l += 8) {
        char* dst = (char*)(vt + (size_t)l * HDIM);
        if (l < len) {
            const int ln = l + 8;
            if (ln < len) {   // prefetch next row for this wave
                const int rn = seqs[b * LSEQ + ln];
                __builtin_prefetch(poi + (size_t)rn * HDIM + lane * 16, 0, 0);
            }
            const float  wl  = wv[l];
            const int    row = seqs[b * LSEQ + l];
            const float4* src = (const float4*)(poi + (size_t)row * HDIM);
#pragma unroll
            for (int j = 0; j < 4; ++j) {
                float4 v = src[lane + 32 * j];
                v4h p;
                p[0] = (_Float16)(v.x * wl); p[1] = (_Float16)(v.y * wl);
                p[2] = (_Float16)(v.z * wl); p[3] = (_Float16)(v.w * wl);
                *(v4h*)(dst + (size_t)(lane + 32 * j) * 8) = p;
            }
        } else {   // w==0 rows: zero-fill, skip the HBM gather entirely
            v4h z = {};
#pragma unroll
            for (int j = 0; j < 4; ++j)
                *(v4h*)(dst + (size_t)(lane + 32 * j) * 8) = z;
        }
    }
    __syncthreads();

    // ---- Phase C: scores[l] = Vt[l,:]·u_node via v_wmma_f32_16x16x32_f16 ---
    // A lane map (16-bit A 16x32): lanes 0-15 row M=lane, K=[0..7]U[16..23];
    // lanes 16-31 row M=lane-16, K=[8..15]U[24..31]. B = u_node broadcast to
    // every column, so D[m][n] = score(row m) for all n.
    for (int t = wid; t < NTILES; t += 8) {
        v8f acc = {};
        const int row  = t * 16 + (lane & 15);
        const _Float16* rowp = vt + (size_t)row * HDIM;
        const int koff = (lane & 16) ? 8 : 0;
#pragma unroll
        for (int kk = 0; kk < 16; ++kk) {
            const int kb = kk * 32 + koff;
            v8h alo = *(const v8h*)(rowp + kb);
            v8h ahi = *(const v8h*)(rowp + kb + 16);
            v8h blo = *(const v8h*)(un + kb);
            v8h bhi = *(const v8h*)(un + kb + 16);
            v16h A  = __builtin_shufflevector(alo, ahi, 0,1,2,3,4,5,6,7,8,9,10,11,12,13,14,15);
            v16h Bm = __builtin_shufflevector(blo, bhi, 0,1,2,3,4,5,6,7,8,9,10,11,12,13,14,15);
            acc = __builtin_amdgcn_wmma_f32_16x16x32_f16(
                      false, A, false, Bm, (short)0, acc, false, false);
        }
        // C/D map: lanes 0-15 hold M=0..7 in acc[0..7]; lanes 16-31 hold M=8..15
        if (lane == 0) {
#pragma unroll
            for (int m = 0; m < 8; ++m) sc[t * 16 + m] = acc[m];
        } else if (lane == 16) {
#pragma unroll
            for (int m = 0; m < 8; ++m) sc[t * 16 + 8 + m] = acc[m];
        }
    }

    // ---- Phase D: s_edge = (sum_l Vt[l,:]/len)·u_edge ----------------------
    {
        const int h1 = tid, h2 = tid + 256;
        float a0 = 0.f, a1 = 0.f;
        for (int l = 0; l < LSEQ; ++l) {
            a0 += (float)vt[(size_t)l * HDIM + h1];
            a1 += (float)vt[(size_t)l * HDIM + h2];
        }
        const float invlen = 1.f / (float)len;
        atomicAdd(&misc[1], (a0 * u_edge[h1] + a1 * u_edge[h2]) * invlen);
    }
    __syncthreads();

    // ---- Phase E: masked leaky-relu softmax (wave 0) -----------------------
    if (wid == 0) {
        const float sE = misc[1];
        float m = -INFINITY;
        for (int l = lane; l < LSEQ; l += 32) {
            float s = sc[l] + sE;
            s = (s > 0.f) ? s : 0.2f * s;
            s = (l < len) ? s : -INFINITY;
            att[l] = s;
            m = fmaxf(m, s);
        }
        for (int off = 16; off; off >>= 1) m = fmaxf(m, __shfl_xor(m, off, 32));
        float sum = 0.f;
        for (int l = lane; l < LPAD; l += 32) {
            float e = (l < len) ? __expf(att[l] - m) : 0.f;
            att[l] = e;
            sum += e;
        }
        for (int off = 16; off; off >>= 1) sum += __shfl_xor(sum, off, 32);
        const float inv = 1.f / sum;
        for (int l = lane; l < LPAD; l += 32) att[l] *= inv;
    }
    __syncthreads();

    // ---- Phase F: out[b,h] = sum_l att[l]*Vt[l,h] --------------------------
    {
        const int h1 = tid, h2 = tid + 256;
        float o0 = 0.f, o1 = 0.f;
        for (int l = 0; l < LSEQ; ++l) {
            const float a = att[l];
            o0 = fmaf(a, (float)vt[(size_t)l * HDIM + h1], o0);
            o1 = fmaf(a, (float)vt[(size_t)l * HDIM + h2], o1);
        }
        out[(size_t)b * HDIM + h1] = o0;
        out[(size_t)b * HDIM + h2] = o1;
    }
}

// ---------------------------------------------------------------------------
extern "C" void kernel_launch(void* const* d_in, const int* in_sizes, int n_in,
                              void* d_out, int out_size, void* d_ws, size_t ws_size,
                              hipStream_t stream)
{
    const float* poi   = (const float*)d_in[0];   // (100000, 512) f32
    const int*   seqs  = (const int*)  d_in[1];   // (2048, 200)   i32
    const int*   lens  = (const int*)  d_in[2];   // (2048,)       i32
    const float* edges = (const float*)d_in[3];   // (2048, 199)   f32
    const float* Wn    = (const float*)d_in[4];   // (512, 512)    f32
    const float* We    = (const float*)d_in[5];   // (512, 512)    f32
    const float* av    = (const float*)d_in[6];   // (1024,)       f32

    float*    u_edge   = (float*)d_ws;                       // 512 f32
    _Float16* u_node_h = (_Float16*)(u_edge + HDIM);         // 512 f16

    hawkes_prep_u<<<2, 256, 0, stream>>>(Wn, We, av, u_edge, u_node_h);

    (void)hipFuncSetAttribute((const void*)hawkes_block_kernel,
                              hipFuncAttributeMaxDynamicSharedMemorySize,
                              SMEM_BYTES);
    hawkes_block_kernel<<<2048, 256, SMEM_BYTES, stream>>>(
        poi, seqs, lens, edges, u_edge, u_node_h, (float*)d_out);
}
